// AttentionBlock_30794915512445
// MI455X (gfx1250) — matrix operand order repaired and checked by reading, not compiled, not run on hardware
//
#include <hip/hip_runtime.h>
#include <hip/hip_bf16.h>
#include <cstdint>

typedef __attribute__((ext_vector_type(16))) __bf16  bf16x16;
typedef __attribute__((ext_vector_type(8)))  float   f32x8;
typedef __attribute__((ext_vector_type(4)))  float   f32x4;
typedef __attribute__((ext_vector_type(4)))  unsigned int u32x4;

#define B_      4
#define C_      256
#define N_      4096
#define CQK_    32
#define NCHUNK  (N_ / 64)
#define VROW    72                 // V row pad: 64 keys -> 72 (144 B, 16B-aligned)

union ABFrag { bf16x16 v; u32x4 u[2]; };
union Pack8  { __bf16 h[8]; u32x4 u; };

// Dynamic-LDS layout for attn_kernel (bytes):
//   Vs[2]   : 2 * 256*72*2 = 73728   (double-buffered V chunk, [chan][key])
//   Es      : 4*16*64*4    = 16384   (per-wave energy tile)
//   Ps      : 4*16*72*2    =  9216   (per-wave bf16 prob tile, padded rows)
//   rowscale: 64*4         =   256
#define VS_BYTES   (C_ * VROW * 2)
#define ES_BYTES   (4 * 16 * 64 * 4)
#define PS_BYTES   (4 * 16 * VROW * 2)
#define SMEM_BYTES (2 * VS_BYTES + ES_BYTES + PS_BYTES + 256)

// ---------------------------------------------------------------------------
// Kernel 1: fused q/k/v 1x1-conv projections, fp32 in -> bf16 workspace.
// ---------------------------------------------------------------------------
__global__ __launch_bounds__(64) void proj_kernel(
    const float* __restrict__ x,
    const float* __restrict__ Wq, const float* __restrict__ bq,
    const float* __restrict__ Wk, const float* __restrict__ bk,
    const float* __restrict__ Wv, const float* __restrict__ bv,
    __bf16* __restrict__ qws, __bf16* __restrict__ kws, __bf16* __restrict__ vws)
{
    __shared__ float xs[C_ * 64];          // 64 KB, XOR-swizzled float4 groups
    const int tid = threadIdx.x;
    const int b   = blockIdx.y;
    const int n0  = blockIdx.x * 64;

    {
        f32x4* xs4 = (f32x4*)xs;
        const f32x4* xsrc = (const f32x4*)(x + ((size_t)b * C_) * N_ + n0);
        for (int i = 0; i < 64; ++i) {
            int idx = tid + i * 64;
            int c = idx >> 4, g = idx & 15;
            f32x4 val = xsrc[(size_t)c * (N_ / 4) + g];
            xs4[c * 16 + (g ^ (c & 15))] = val;
        }
    }
    __syncthreads();

    const int n  = tid;
    const int ng = n0 + n;

    const float* Wqk[2] = { Wq, Wk };
    const float* bqk[2] = { bq, bk };
    __bf16*      oqk[2] = { qws, kws };
    for (int m = 0; m < 2; ++m) {
        const float* W = Wqk[m];
        for (int ob = 0; ob < CQK_; ob += 8) {
            float acc[8] = {0,0,0,0,0,0,0,0};
            for (int c = 0; c < C_; ++c) {
                float xv = xs[c * 64 + ((((n >> 2) ^ (c & 15)) << 2) | (n & 3))];
                #pragma unroll
                for (int j = 0; j < 8; ++j)
                    acc[j] = fmaf(W[(ob + j) * C_ + c], xv, acc[j]);
            }
            Pack8 pk;
            #pragma unroll
            for (int j = 0; j < 8; ++j)
                pk.h[j] = (__bf16)(acc[j] + bqk[m][ob + j]);
            *(u32x4*)(oqk[m] + ((size_t)(b * N_ + ng)) * CQK_ + ob) = pk.u;
        }
    }
    for (int ob = 0; ob < C_; ob += 8) {
        float acc[8] = {0,0,0,0,0,0,0,0};
        for (int c = 0; c < C_; ++c) {
            float xv = xs[c * 64 + ((((n >> 2) ^ (c & 15)) << 2) | (n & 3))];
            #pragma unroll
            for (int j = 0; j < 8; ++j)
                acc[j] = fmaf(Wv[(ob + j) * C_ + c], xv, acc[j]);
        }
        #pragma unroll
        for (int j = 0; j < 8; ++j)
            vws[((size_t)(b * C_ + ob + j)) * N_ + ng] = (__bf16)(acc[j] + bv[ob + j]);
    }
}

// ---------------------------------------------------------------------------
// Async stage of one 64-key V chunk [256 chans][64 keys] into an LDS buffer
// using GLOBAL_LOAD_ASYNC_TO_LDS_B128 (ASYNCcnt-tracked, no VGPR bounce).
// 128 threads x 16 issues x 16 B/lane = 32 KB.
// ---------------------------------------------------------------------------
__device__ __forceinline__ void stage_v_async(const __bf16* vbase, int kc,
                                              __bf16* vbuf, int tid)
{
    #pragma unroll
    for (int i = 0; i < 16; ++i) {
        int idx = tid + i * 128;
        int c = idx >> 3, g = idx & 7;
        unsigned gb = (unsigned)((c * N_ + kc * 64 + g * 8) * 2);       // byte off
        unsigned lb = (unsigned)(uintptr_t)(vbuf + c * VROW + g * 8);   // LDS off
        asm volatile("global_load_async_to_lds_b128 %0, %1, %2 offset:0"
                     :: "v"(lb), "v"(gb), "s"(vbase) : "memory");
    }
}

// ---------------------------------------------------------------------------
// Kernel 2: streaming (flash) attention + residual, double-buffered async V.
// Grid (N/64, B), block 128 (4 waves, 16 queries/wave).
// ---------------------------------------------------------------------------
__global__ __launch_bounds__(128) void attn_kernel(
    const __bf16* __restrict__ qws, const __bf16* __restrict__ kws,
    const __bf16* __restrict__ vws, const float* __restrict__ xin,
    const float* __restrict__ gamma_p, float* __restrict__ out)
{
    extern __shared__ char smem[];
    __bf16* Vs0 = (__bf16*)smem;
    __bf16* Vs1 = Vs0 + C_ * VROW;
    float*  Es  = (float*)(smem + 2 * VS_BYTES);
    __bf16* Ps  = (__bf16*)(smem + 2 * VS_BYTES + ES_BYTES);
    float*  rowscale = (float*)(smem + 2 * VS_BYTES + ES_BYTES + PS_BYTES);

    const int tid  = threadIdx.x;
    const int w    = tid >> 5;
    const int lane = tid & 31;
    const int hh   = lane >> 4;
    const int ln   = lane & 15;
    const int b    = blockIdx.y;
    const int q0   = blockIdx.x * 64 + w * 16;

    const __bf16* vbase = vws + ((size_t)b * C_) * N_;

    // Resident Q A-fragment (ISA 16-bit A layout: half 0 K in {0-7,16-23},
    // half 1 K in {8-15,24-31}).
    ABFrag qa;
    {
        const __bf16* qp = qws + ((size_t)(b * N_ + q0 + ln)) * CQK_;
        qa.u[0] = *(const u32x4*)(qp + 8 * hh);
        qa.u[1] = *(const u32x4*)(qp + 16 + 8 * hh);
    }

    f32x8 acc[16];
    #pragma unroll
    for (int ct = 0; ct < 16; ++ct)
        acc[ct] = f32x8{0.f,0.f,0.f,0.f,0.f,0.f,0.f,0.f};

    float m_run = -__builtin_inff();
    float l_run = 0.f;

    stage_v_async(vbase, 0, Vs0, tid);

    for (int kc = 0; kc < NCHUNK; ++kc) {
        __bf16* vb = (kc & 1) ? Vs1 : Vs0;
        if (kc + 1 < NCHUNK) {
            stage_v_async(vbase, kc + 1, (kc & 1) ? Vs0 : Vs1, tid);
            asm volatile("s_wait_asynccnt 0x10" ::: "memory"); // current chunk landed
        } else {
            asm volatile("s_wait_asynccnt 0x0" ::: "memory");
        }
        __syncthreads();

        // ---- energy: 16 queries x 64 keys, 4 bf16 WMMAs ----
        for (int kb = 0; kb < 4; ++kb) {
            const __bf16* kp =
                kws + ((size_t)(b * N_ + kc * 64 + kb * 16 + ln)) * CQK_ + 8 * hh;
            ABFrag kf;
            kf.u[0] = *(const u32x4*)(kp);
            kf.u[1] = *(const u32x4*)(kp + 16);
            f32x8 e = f32x8{0.f,0.f,0.f,0.f,0.f,0.f,0.f,0.f};
            e = __builtin_amdgcn_wmma_f32_16x16x32_bf16(
                    false, qa.v, false, kf.v, (short)0, e, false, false);
            #pragma unroll
            for (int r = 0; r < 8; ++r)
                Es[(w * 16 + r + 8 * hh) * 64 + kb * 16 + ln] = e[r];
        }
        __builtin_amdgcn_wave_barrier();

        // ---- online softmax: row = ln, each half-wave scans 32 keys ----
        {
            const float* erow = &Es[(w * 16 + ln) * 64 + 32 * hh];
            float cm = -__builtin_inff();
            for (int j = 0; j < 32; ++j) cm = fmaxf(cm, erow[j]);
            cm = fmaxf(cm, __shfl_xor(cm, 16));
            float nm = fmaxf(m_run, cm);
            float sc = __expf(m_run - nm);
            float s = 0.f;
            __bf16* prow = &Ps[(w * 16 + ln) * VROW + 32 * hh];
            for (int j = 0; j < 32; ++j) {
                float p = __expf(erow[j] - nm);
                s += p;
                prow[j] = (__bf16)p;
            }
            s += __shfl_xor(s, 16);
            m_run = nm;
            l_run = l_run * sc + s;
            if (lane < 16) rowscale[w * 16 + ln] = sc;
        }
        __builtin_amdgcn_wave_barrier();

        // ---- rescale accumulators ----
        float rs[8];
        #pragma unroll
        for (int r = 0; r < 8; ++r) rs[r] = rowscale[w * 16 + r + 8 * hh];
        #pragma unroll
        for (int ct = 0; ct < 16; ++ct)
            #pragma unroll
            for (int r = 0; r < 8; ++r) acc[ct][r] *= rs[r];

        // ---- out += P(16x64) x V(64x16 per channel tile): 32 bf16 WMMAs ----
        ABFrag pa[2];
        {
            const __bf16* prow = &Ps[(w * 16 + ln) * VROW];
            #pragma unroll
            for (int kh = 0; kh < 2; ++kh) {
                pa[kh].u[0] = *(const u32x4*)(prow + 32 * kh + 8 * hh);
                pa[kh].u[1] = *(const u32x4*)(prow + 32 * kh + 16 + 8 * hh);
            }
        }
        #pragma unroll
        for (int ct = 0; ct < 16; ++ct) {
            const __bf16* vrow = &vb[(ct * 16 + ln) * VROW];
            #pragma unroll
            for (int kh = 0; kh < 2; ++kh) {
                ABFrag vf;
                vf.u[0] = *(const u32x4*)(vrow + 32 * kh + 8 * hh);
                vf.u[1] = *(const u32x4*)(vrow + 32 * kh + 16 + 8 * hh);
                acc[ct] = __builtin_amdgcn_wmma_f32_16x16x32_bf16(
                              false, pa[kh].v, false, vf.v, (short)0, acc[ct],
                              false, false);
            }
        }
        __syncthreads();   // all waves done with vb before it is overwritten
    }

    // ---- finalize: 1/l, transpose via LDS (reuse V buffers), coalesced IO ----
    if (lane < 16) rowscale[w * 16 + ln] = 1.0f / l_run;
    __builtin_amdgcn_wave_barrier();

    const float gamma = gamma_p[0];
    float inv[8];
    #pragma unroll
    for (int r = 0; r < 8; ++r) inv[r] = rowscale[w * 16 + r + 8 * hh];

    float* tb = (float*)smem;            // per-wave 64 rows x 20 floats (padded)
    const int wbase = w * 1280;
    for (int p = 0; p < 4; ++p) {
        #pragma unroll
        for (int ctl = 0; ctl < 4; ++ctl) {
            const int ct = p * 4 + ctl;
            #pragma unroll
            for (int r = 0; r < 8; ++r)
                tb[wbase + (ctl * 16 + ln) * 20 + r + 8 * hh] = acc[ct][r] * inv[r];
        }
        __builtin_amdgcn_wave_barrier();
        #pragma unroll
        for (int rr = 0; rr < 2; ++rr) {
            const int row  = lane * 2 + rr;          // local chan 0..63
            const int chan = p * 64 + row;
            const size_t base = ((size_t)(b * C_ + chan)) * N_ + q0;
            #pragma unroll
            for (int seg = 0; seg < 4; ++seg) {
                f32x4 v  = *(const f32x4*)&tb[wbase + row * 20 + seg * 4];
                f32x4 xv = *(const f32x4*)&xin[base + seg * 4];
                f32x4 res = v * gamma + xv;
                *(f32x4*)&out[base + seg * 4] = res;
            }
        }
        __builtin_amdgcn_wave_barrier();
    }
}

// ---------------------------------------------------------------------------
extern "C" void kernel_launch(void* const* d_in, const int* in_sizes, int n_in,
                              void* d_out, int out_size, void* d_ws, size_t ws_size,
                              hipStream_t stream) {
    const float* x     = (const float*)d_in[0];
    const float* Wq    = (const float*)d_in[1];
    const float* bq    = (const float*)d_in[2];
    const float* Wk    = (const float*)d_in[3];
    const float* bk    = (const float*)d_in[4];
    const float* Wv    = (const float*)d_in[5];
    const float* bv    = (const float*)d_in[6];
    const float* gamma = (const float*)d_in[7];

    char* ws = (char*)d_ws;
    __bf16* qws = (__bf16*)ws;                                       // 1 MB
    __bf16* kws = (__bf16*)(ws + (size_t)B_ * N_ * CQK_ * 2);        // 1 MB
    __bf16* vws = (__bf16*)(ws + (size_t)B_ * N_ * CQK_ * 4);        // 8 MB

    proj_kernel<<<dim3(N_ / 64, B_), 64, 0, stream>>>(
        x, Wq, bq, Wk, bk, Wv, bv, qws, kws, vws);
    attn_kernel<<<dim3(N_ / 64, B_), 128, SMEM_BYTES, stream>>>(
        qws, kws, vws, x, gamma, (float*)d_out);
}
